// ColBERT_69423851373265
// MI455X (gfx1250) — compile-verified, matrix operand order checked by reading, not run on hardware
//
#include <hip/hip_runtime.h>

// ColBERT MaxSim on MI455X (gfx1250, wave32).
// 17.2 GFLOP vs 134 MB traffic (AI ~128 FLOP/B) -> use v_wmma_f32_16x16x32_f16.
// One workgroup per doc n: normalize doc tile fp32->f16 into 64KB LDS once,
// reuse across all 8 batches; q normalized once into workspace (L2-resident).

typedef _Float16 half4  __attribute__((ext_vector_type(4)));
typedef _Float16 half8  __attribute__((ext_vector_type(8)));
typedef _Float16 half16 __attribute__((ext_vector_type(16)));
typedef float    v8f    __attribute__((ext_vector_type(8)));

#define B_   8
#define LQ   32
#define N_   1024
#define LD   256
#define K_   128
#define NEGV -100.0f
#define EPSV 1e-8f

// ---- Kernel 1: normalize q (256 rows of 128 fp32) -> f16 in workspace ----
__global__ void __launch_bounds__(256) qnorm_kernel(const float* __restrict__ q,
                                                    _Float16* __restrict__ qn) {
    const int r = threadIdx.x;                 // 256 rows, one block of 256
    const float4* src = (const float4*)(q + (size_t)r * K_);
    float s = 0.f;
#pragma unroll
    for (int i = 0; i < K_ / 4; ++i) {
        float4 v = src[i];
        s += v.x * v.x + v.y * v.y + v.z * v.z + v.w * v.w;
    }
    const float inv = 1.0f / fmaxf(sqrtf(s), EPSV);
    _Float16* dst = qn + (size_t)r * K_;
#pragma unroll
    for (int i = 0; i < K_ / 4; ++i) {
        float4 v = src[i];
        half4 h;
        h[0] = (_Float16)(v.x * inv);
        h[1] = (_Float16)(v.y * inv);
        h[2] = (_Float16)(v.z * inv);
        h[3] = (_Float16)(v.w * inv);
        *(half4*)(dst + i * 4) = h;
    }
}

static __device__ inline half16 mk16(half8 lo, half8 hi) {
    return __builtin_shufflevector(lo, hi, 0, 1, 2, 3, 4, 5, 6, 7,
                                   8, 9, 10, 11, 12, 13, 14, 15);
}

// ---- Kernel 2: one workgroup per doc n; 8 waves; WMMA f16 -> f32 ----
__global__ void __launch_bounds__(256) colbert_maxsim_kernel(
    const _Float16* __restrict__ qn, const int* __restrict__ q_mask,
    const float* __restrict__ d_hidden, const int* __restrict__ d_mask,
    float* __restrict__ out) {
    extern __shared__ _Float16 sd[];           // [LD][K_] f16 doc tile, 64 KB
    __shared__ float spartial[8][32];          // per-wave row-max partials

    const int n     = blockIdx.x;
    const int tid   = threadIdx.x;
    const int w     = tid >> 5;                // wave id 0..7 (wave32)
    const int lane  = tid & 31;
    const int lmod  = lane & 15;
    const int halfg = lane >> 4;               // 16-lane half-group

    // ---- Stage doc n: normalize fp32 -> f16 into LDS (each thread = 1 row)
    {
        const int row = tid;                   // 256 rows, 256 threads
        const float4* src =
            (const float4*)(d_hidden + ((size_t)n * LD + row) * K_);
        float s = 0.f;
#pragma unroll
        for (int i = 0; i < K_ / 4; ++i) {
            float4 v = src[i];
            s += v.x * v.x + v.y * v.y + v.z * v.z + v.w * v.w;
        }
        const float inv = 1.0f / fmaxf(sqrtf(s), EPSV);
        _Float16* dst = sd + (size_t)row * K_;
#pragma unroll
        for (int i = 0; i < K_ / 4; ++i) {
            float4 v = src[i];
            half4 h;
            h[0] = (_Float16)(v.x * inv);
            h[1] = (_Float16)(v.y * inv);
            h[2] = (_Float16)(v.z * inv);
            h[3] = (_Float16)(v.w * inv);
            *(half4*)(dst + i * 4) = h;
        }
    }
    __syncthreads();

    // Each wave owns two 16-column tiles of the 256 doc tokens.
    const int col0 = (w * 2 + 0) * 16 + lmod;
    const int col1 = (w * 2 + 1) * 16 + lmod;
    const bool dm0 = d_mask[n * LD + col0] > 0;
    const bool dm1 = d_mask[n * LD + col1] > 0;

    for (int b = 0; b < B_; ++b) {
        v8f acc[2][2] = {};                    // [ct_local][m_tile]
#pragma unroll
        for (int kc = 0; kc < 4; ++kc) {
            const int kk   = kc * 32;
            const int offK = kk + 8 * halfg;   // ISA 16-bit A layout: K interleave

            // A fragments (q rows), from global f16 qn (hot in L2)
            const _Float16* qa0 = qn + (size_t)(b * LQ + lmod) * K_;
            const _Float16* qa1 = qn + (size_t)(b * LQ + 16 + lmod) * K_;
            half16 A0 = mk16(*(const half8*)(qa0 + offK),
                             *(const half8*)(qa0 + offK + 16));
            half16 A1 = mk16(*(const half8*)(qa1 + offK),
                             *(const half8*)(qa1 + offK + 16));

            // B fragments (doc columns) from LDS, symmetric K interleave
            const _Float16* db0 = sd + (size_t)col0 * K_ + offK;
            const _Float16* db1 = sd + (size_t)col1 * K_ + offK;
            half16 Bf0 = mk16(*(const half8*)(db0), *(const half8*)(db0 + 16));
            half16 Bf1 = mk16(*(const half8*)(db1), *(const half8*)(db1 + 16));

            acc[0][0] = __builtin_amdgcn_wmma_f32_16x16x32_f16(
                false, A0, false, Bf0, (short)0, acc[0][0], false, false);
            acc[0][1] = __builtin_amdgcn_wmma_f32_16x16x32_f16(
                false, A1, false, Bf0, (short)0, acc[0][1], false, false);
            acc[1][0] = __builtin_amdgcn_wmma_f32_16x16x32_f16(
                false, A0, false, Bf1, (short)0, acc[1][0], false, false);
            acc[1][1] = __builtin_amdgcn_wmma_f32_16x16x32_f16(
                false, A1, false, Bf1, (short)0, acc[1][1], false, false);
        }

        // Masked max over this wave's 32 doc columns, per q row.
        // C/D layout: lane L, VGPR v -> (M = v + 8*(L/16), N = L%16).
#pragma unroll
        for (int mt = 0; mt < 2; ++mt) {
#pragma unroll
            for (int v = 0; v < 8; ++v) {
                float x0 = dm0 ? acc[0][mt][v] : NEGV;
                float x1 = dm1 ? acc[1][mt][v] : NEGV;
                float x  = fmaxf(x0, x1);
                // reduce across the 16 lanes of this half-group (columns)
                x = fmaxf(x, __shfl_xor(x, 1, 32));
                x = fmaxf(x, __shfl_xor(x, 2, 32));
                x = fmaxf(x, __shfl_xor(x, 4, 32));
                x = fmaxf(x, __shfl_xor(x, 8, 32));
                if (lmod == 0)
                    spartial[w][mt * 16 + 8 * halfg + v] = x;
            }
        }
        __syncthreads();

        // Wave 0: combine 8 wave partials, apply q_mask, sum 32 rows.
        if (tid < 32) {
            float m = spartial[0][tid];
#pragma unroll
            for (int ww = 1; ww < 8; ++ww) m = fmaxf(m, spartial[ww][tid]);
            float val = (q_mask[b * LQ + tid] > 0) ? m : 0.0f;
            val += __shfl_xor(val, 16, 32);
            val += __shfl_xor(val, 8, 32);
            val += __shfl_xor(val, 4, 32);
            val += __shfl_xor(val, 2, 32);
            val += __shfl_xor(val, 1, 32);
            if (tid == 0) out[b * N_ + n] = val;
        }
        __syncthreads();   // spartial reused next b
    }
}

extern "C" void kernel_launch(void* const* d_in, const int* in_sizes, int n_in,
                              void* d_out, int out_size, void* d_ws, size_t ws_size,
                              hipStream_t stream) {
    const float* q_hidden = (const float*)d_in[0];
    const int*   q_mask   = (const int*)d_in[1];
    const float* d_hidden = (const float*)d_in[2];
    const int*   d_mask   = (const int*)d_in[3];
    float*       out      = (float*)d_out;

    _Float16* qn = (_Float16*)d_ws;            // 8*32*128 f16 = 64 KB

    qnorm_kernel<<<1, 256, 0, stream>>>(q_hidden, qn);

    const size_t shbytes = (size_t)LD * K_ * sizeof(_Float16);  // 64 KB LDS
    colbert_maxsim_kernel<<<N_, 256, shbytes, stream>>>(
        qn, q_mask, d_hidden, d_mask, out);
}
